// ContextAwareMoEGate_54546084659702
// MI455X (gfx1250) — matrix operand order repaired and checked by reading, not compiled, not run on hardware
//
#include <hip/hip_runtime.h>
#include <hip/hip_bf16.h>
#include <math.h>

typedef __attribute__((ext_vector_type(16))) _Float16 v16h;
typedef __attribute__((ext_vector_type(8)))  float    v8f;

#define B_SZ   16384
#define T_SZ   7
#define D_SZ   1024
#define H_SZ   512
#define E_SZ   64
#define TOPK_K 8
#define NFLOW  4
#define G4H    2048                     // 4*H
#define KW_GW1 (D_SZ + H_SZ + NFLOW)    // 1540, row stride of original gw1
#define KW_G16 (D_SZ + H_SZ)            // 1536, row stride of converted gw1

// ---------------------------------------------------------------------------
// WMMA fragment helpers (layouts per CDNA5 ISA 7.12.2, 16-bit operands)
// A operand (16xK slab): lane l -> row = l&15, half h = l>>4;
//   elems 0..7  = K kbase + h*8 + i ; elems 8..15 = K kbase + 16 + h*8 + i
// B operand (Kx16, loaded as rows of W for C = A * W^T): lane l -> col = l&15,
//   h = l>>4; elems 0..7 = K kbase + h*8 + i ; elems 8..15 = K kbase + 16 + h*8 + i
// ---------------------------------------------------------------------------
static __device__ __forceinline__ v16h load_a_f32(const float* __restrict__ src,
                                                  long ld, int row, int kbase,
                                                  int lane) {
  const int r = lane & 15, h = lane >> 4;
  const float* p0 = src + (long)(row + r) * ld + (kbase + h * 8);
  const float* p1 = p0 + 16;
  v16h o;
#pragma unroll
  for (int i = 0; i < 8; ++i) {
    o[i]     = (_Float16)p0[i];
    o[i + 8] = (_Float16)p1[i];
  }
  return o;
}

static __device__ __forceinline__ v16h load_a_f16(const _Float16* __restrict__ src,
                                                  long ld, int row, int kbase,
                                                  int lane) {
  const int r = lane & 15, h = lane >> 4;
  const _Float16* p0 = src + (long)(row + r) * ld + (kbase + h * 8);
  const _Float16* p1 = p0 + 16;
  v16h o;
#pragma unroll
  for (int i = 0; i < 8; ++i) {
    o[i]     = p0[i];
    o[i + 8] = p1[i];
  }
  return o;
}

static __device__ __forceinline__ v16h load_b_f16(const _Float16* __restrict__ w,
                                                  long ld, int col, int kbase,
                                                  int lane) {
  const int n = lane & 15, h = lane >> 4;
  const _Float16* p = w + (long)(col + n) * ld + (kbase + h * 8);
  v16h o;
#pragma unroll
  for (int i = 0; i < 8; ++i) {
    o[i]     = p[i];
    o[i + 8] = p[i + 16];
  }
  return o;
}

static __device__ __forceinline__ v16h load_a_lds(const _Float16* s, int ld,
                                                  int kbase, int lane) {
  const int r = lane & 15, h = lane >> 4;
  const _Float16* p0 = s + r * ld + kbase + h * 8;
  const _Float16* p1 = p0 + 16;
  v16h o;
#pragma unroll
  for (int i = 0; i < 8; ++i) {
    o[i]     = p0[i];
    o[i + 8] = p1[i];
  }
  return o;
}

static __device__ __forceinline__ v8f wmma_f16(v16h a, v16h b, v8f c) {
  // 8 args: (neg_a, A, neg_b, B, c_mod, C, reuse_a, reuse_b)
  return __builtin_amdgcn_wmma_f32_16x16x32_f16(false, a, false, b, (short)0, c,
                                                false, false);
}

static __device__ __forceinline__ float sigm(float x) {
  return 1.0f / (1.0f + __expf(-x));
}

// ---------------------------------------------------------------------------
__global__ void zero_kernel(float* __restrict__ p, unsigned long n) {
  unsigned long i = (unsigned long)blockIdx.x * blockDim.x + threadIdx.x;
  if (i < n) p[i] = 0.0f;
}

// fp32 -> f16 weight conversion with independent src/dst row strides
__global__ void cvt_kernel(const float* __restrict__ src,
                           _Float16* __restrict__ dst, long src_ld, long dst_ld,
                           long cols, long total) {
  long i = (long)blockIdx.x * blockDim.x + threadIdx.x;
  if (i >= total) return;
  long r = i / cols, c = i - r * cols;
  dst[r * dst_ld + c] = (_Float16)src[r * src_ld + c];
}

// ---------------------------------------------------------------------------
// One LSTM timestep: gates = ctx[:,t,:]@W_ih^T + h@W_hh^T (+b), then cell.
// Workgroup = 32 batch rows (2 row-tiles), 8 waves; each wave computes a
// 32x128 C-slab (2x8 WMMA accumulators) so every B-frag is reused across two
// row tiles (halves weight traffic vs 16-row blocks). B-frags are prefetched
// one step ahead to keep loads in flight under the WMMAs. Gate epilogue +
// fused cell run in two row-tile passes through a single 64 KB LDS buffer.
// ---------------------------------------------------------------------------
__global__ __launch_bounds__(256) void lstm_step_kernel(
    const float* __restrict__ ctx, const _Float16* __restrict__ wih16,
    const _Float16* __restrict__ whh16, const float* __restrict__ b_ih,
    const float* __restrict__ b_hh, _Float16* __restrict__ h16,
    float* __restrict__ c_buf, int t) {
  __shared__ _Float16 gates[16][G4H];   // 64 KB, reused for both row tiles
  const int tid  = threadIdx.x;
  const int lane = tid & 31;
  const int wave = tid >> 5;
  const int m0   = blockIdx.x * 32;
  const int nB   = wave * 128;

  const v8f vzero = {};
  v8f acc0[8], acc1[8];
#pragma unroll
  for (int i = 0; i < 8; ++i) { acc0[i] = vzero; acc1[i] = vzero; }

  // K over x: context[:, t, :], row stride T*D (fp32 -> f16 on the fly)
  for (int ks = 0; ks < D_SZ / 32; ++ks) {
    const int kb = ks * 32;
    v16h a0 = load_a_f32(ctx + (long)t * D_SZ, (long)T_SZ * D_SZ, m0, kb, lane);
    v16h a1 = load_a_f32(ctx + (long)t * D_SZ, (long)T_SZ * D_SZ, m0 + 16, kb, lane);
    v16h b = load_b_f16(wih16, D_SZ, nB, kb, lane);
#pragma unroll
    for (int nt = 0; nt < 8; ++nt) {
      v16h bn = b;
      if (nt < 7) bn = load_b_f16(wih16, D_SZ, nB + (nt + 1) * 16, kb, lane);
      acc0[nt] = wmma_f16(a0, b, acc0[nt]);
      acc1[nt] = wmma_f16(a1, b, acc1[nt]);
      b = bn;
    }
  }
  // K over h_prev (already f16)
  for (int ks = 0; ks < H_SZ / 32; ++ks) {
    const int kb = ks * 32;
    v16h a0 = load_a_f16(h16, H_SZ, m0, kb, lane);
    v16h a1 = load_a_f16(h16, H_SZ, m0 + 16, kb, lane);
    v16h b = load_b_f16(whh16, H_SZ, nB, kb, lane);
#pragma unroll
    for (int nt = 0; nt < 8; ++nt) {
      v16h bn = b;
      if (nt < 7) bn = load_b_f16(whh16, H_SZ, nB + (nt + 1) * 16, kb, lane);
      acc0[nt] = wmma_f16(a0, b, acc0[nt]);
      acc1[nt] = wmma_f16(a1, b, acc1[nt]);
      b = bn;
    }
  }

  // two passes: stage row-tile accumulators -> LDS, then fused LSTM cell
#pragma unroll
  for (int pass = 0; pass < 2; ++pass) {
    const int mbase = m0 + pass * 16;
#pragma unroll
    for (int nt = 0; nt < 8; ++nt) {
      const int n  = nB + nt * 16 + (lane & 15);
      const int mr = (lane >> 4) * 8;
#pragma unroll
      for (int v = 0; v < 8; ++v)
        gates[mr + v][n] = (_Float16)(pass == 0 ? acc0[nt][v] : acc1[nt][v]);
    }
    __syncthreads();
    // fused LSTM cell: 16 rows x 512 hidden = 8192 elems / 256 threads
    for (int it = 0; it < (16 * H_SZ) / 256; ++it) {
      const int idx = tid + it * 256;
      const int r   = idx >> 9;             // / 512
      const int hh  = idx & (H_SZ - 1);
      const long gm = (long)(mbase + r) * H_SZ + hh;
      float gi = (float)gates[r][hh]            + b_ih[hh]            + b_hh[hh];
      float gf = (float)gates[r][H_SZ + hh]     + b_ih[H_SZ + hh]     + b_hh[H_SZ + hh];
      float gg = (float)gates[r][2 * H_SZ + hh] + b_ih[2 * H_SZ + hh] + b_hh[2 * H_SZ + hh];
      float go = (float)gates[r][3 * H_SZ + hh] + b_ih[3 * H_SZ + hh] + b_hh[3 * H_SZ + hh];
      float c  = sigm(gf) * c_buf[gm] + sigm(gi) * tanhf(gg);
      float h  = sigm(go) * tanhf(c);
      c_buf[gm] = c;
      h16[gm]   = (_Float16)h;
    }
    __syncthreads();
  }
}

// ---------------------------------------------------------------------------
// flow_state = softmax(relu(x@fw1^T + fb1) @ fw2^T + fb2)
// ---------------------------------------------------------------------------
__global__ __launch_bounds__(256) void flow_kernel(
    const float* __restrict__ x, const _Float16* __restrict__ fw116,
    const float* __restrict__ fb1, const float* __restrict__ fw2,
    const float* __restrict__ fb2, float* __restrict__ flow_ws,
    float* __restrict__ flow_out) {
  __shared__ _Float16 hid[16][H_SZ];    // 16 KB
  const int tid  = threadIdx.x;
  const int lane = tid & 31;
  const int wave = tid >> 5;
  const int m0   = blockIdx.x * 16;
  const int nB   = wave * 64;

  const v8f vzero = {};
  v8f acc[4];
#pragma unroll
  for (int i = 0; i < 4; ++i) acc[i] = vzero;

  for (int ks = 0; ks < D_SZ / 32; ++ks) {
    const int kb = ks * 32;
    v16h a = load_a_f32(x, D_SZ, m0, kb, lane);
#pragma unroll
    for (int nt = 0; nt < 4; ++nt) {
      v16h b = load_b_f16(fw116, D_SZ, nB + nt * 16, kb, lane);
      acc[nt] = wmma_f16(a, b, acc[nt]);
    }
  }
#pragma unroll
  for (int nt = 0; nt < 4; ++nt) {
    const int n  = nB + nt * 16 + (lane & 15);
    const int mr = (lane >> 4) * 8;
#pragma unroll
    for (int v = 0; v < 8; ++v) {
      float val = acc[nt][v] + fb1[n];
      hid[mr + v][n] = (_Float16)fmaxf(val, 0.0f);
    }
  }
  __syncthreads();

  if (tid < 16) {
    const int r = tid;
    float z[NFLOW];
#pragma unroll
    for (int j = 0; j < NFLOW; ++j) {
      float s = fb2[j];
      for (int k = 0; k < H_SZ; ++k) s += (float)hid[r][k] * fw2[j * H_SZ + k];
      z[j] = s;
    }
    float mx = fmaxf(fmaxf(z[0], z[1]), fmaxf(z[2], z[3]));
    float den = 0.0f;
#pragma unroll
    for (int j = 0; j < NFLOW; ++j) { z[j] = __expf(z[j] - mx); den += z[j]; }
    const float inv = 1.0f / den;
    const long  m   = m0 + r;
#pragma unroll
    for (int j = 0; j < NFLOW; ++j) {
      float p = z[j] * inv;
      flow_ws[m * NFLOW + j]  = p;
      flow_out[m * NFLOW + j] = p;
    }
  }
}

// ---------------------------------------------------------------------------
// Main gate: hidden = relu([x|h|flow]@gw1^T + gb1) (WMMA, flow cols as rank-4
// epilogue), logits = hidden@gw2^T + gb2 (WMMA, A from LDS), then adjusted
// logits, softmax(64), top-8, usage accumulation.
// ---------------------------------------------------------------------------
__global__ __launch_bounds__(256) void gate_kernel(
    const float* __restrict__ x, const _Float16* __restrict__ h16,
    const float* __restrict__ flow_ws, const _Float16* __restrict__ gw116,
    const float* __restrict__ gw1, const float* __restrict__ gb1,
    const _Float16* __restrict__ gw216, const float* __restrict__ gb2,
    const float* __restrict__ spec, float* __restrict__ usage_acc,
    float* __restrict__ out_topkp, int* __restrict__ out_topki,
    float* __restrict__ out_adj) {
  __shared__ _Float16 hid[16][D_SZ];    // 32 KB
  __shared__ float    lg[16][E_SZ];     // 4 KB
  __shared__ float    us[E_SZ];
  const int tid  = threadIdx.x;
  const int lane = tid & 31;
  const int wave = tid >> 5;
  const int m0   = blockIdx.x * 16;
  const int nB   = wave * 128;

  if (tid < E_SZ) us[tid] = 0.0f;

  const v8f vzero = {};
  v8f acc[8];
#pragma unroll
  for (int i = 0; i < 8; ++i) acc[i] = vzero;

  // K over x part (combined cols 0..1023)
  for (int ks = 0; ks < D_SZ / 32; ++ks) {
    const int kb = ks * 32;
    v16h a = load_a_f32(x, D_SZ, m0, kb, lane);
    v16h b = load_b_f16(gw116, KW_G16, nB, kb, lane);
#pragma unroll
    for (int nt = 0; nt < 8; ++nt) {
      v16h bn = b;
      if (nt < 7) bn = load_b_f16(gw116, KW_G16, nB + (nt + 1) * 16, kb, lane);
      acc[nt] = wmma_f16(a, b, acc[nt]);
      b = bn;
    }
  }
  // K over context_feature part (combined cols 1024..1535), h already f16
  for (int ks = 0; ks < H_SZ / 32; ++ks) {
    const int kb = ks * 32;
    v16h a = load_a_f16(h16, H_SZ, m0, kb, lane);
    v16h b = load_b_f16(gw116, KW_G16, nB, D_SZ + kb, lane);
#pragma unroll
    for (int nt = 0; nt < 8; ++nt) {
      v16h bn = b;
      if (nt < 7)
        bn = load_b_f16(gw116, KW_G16, nB + (nt + 1) * 16, D_SZ + kb, lane);
      acc[nt] = wmma_f16(a, b, acc[nt]);
      b = bn;
    }
  }
  // epilogue: + bias + rank-4 flow term, ReLU, stage f16 to LDS
#pragma unroll
  for (int nt = 0; nt < 8; ++nt) {
    const int n  = nB + nt * 16 + (lane & 15);
    const int mr = (lane >> 4) * 8;
    float w4[NFLOW];
#pragma unroll
    for (int j = 0; j < NFLOW; ++j) w4[j] = gw1[(long)n * KW_GW1 + D_SZ + H_SZ + j];
    const float bn = gb1[n];
#pragma unroll
    for (int v = 0; v < 8; ++v) {
      const long m = m0 + mr + v;
      float fs = 0.0f;
#pragma unroll
      for (int j = 0; j < NFLOW; ++j) fs += flow_ws[m * NFLOW + j] * w4[j];
      float val = acc[nt][v] + bn + fs;
      hid[mr + v][n] = (_Float16)fmaxf(val, 0.0f);
    }
  }
  __syncthreads();

  // second GEMM: [16x1024] (LDS, f16) @ gw2^T -> [16x64]; waves 0..3
  if (wave < 4) {
    v8f lacc = vzero;
    for (int ks = 0; ks < D_SZ / 32; ++ks) {
      const int kb = ks * 32;
      v16h a = load_a_lds(&hid[0][0], D_SZ, kb, lane);
      v16h b = load_b_f16(gw216, D_SZ, wave * 16, kb, lane);
      lacc = wmma_f16(a, b, lacc);
    }
    const int n  = wave * 16 + (lane & 15);
    const int mr = (lane >> 4) * 8;
#pragma unroll
    for (int v = 0; v < 8; ++v) lg[mr + v][n] = lacc[v] + gb2[n];
  }
  __syncthreads();

  // per-row: adjust, softmax(64), usage, top-8
  if (tid < 16) {
    const int r = tid;
    const long m = m0 + r;
    float fl[NFLOW];
#pragma unroll
    for (int j = 0; j < NFLOW; ++j) fl[j] = flow_ws[m * NFLOW + j];
    float mx = -3.0e38f;
    for (int e = 0; e < E_SZ; ++e) {
      float sp = spec[e * NFLOW + 0] * fl[0] + spec[e * NFLOW + 1] * fl[1] +
                 spec[e * NFLOW + 2] * fl[2] + spec[e * NFLOW + 3] * fl[3];
      float adj = lg[r][e] + 0.1f * sp;
      lg[r][e] = adj;
      out_adj[m * E_SZ + e] = adj;
      mx = fmaxf(mx, adj);
    }
    float den = 0.0f;
    for (int e = 0; e < E_SZ; ++e) {
      float p = __expf(lg[r][e] - mx);
      lg[r][e] = p;
      den += p;
    }
    const float inv = 1.0f / den;
    for (int e = 0; e < E_SZ; ++e) {
      float p = lg[r][e] * inv;
      lg[r][e] = p;
      atomicAdd(&us[e], p);   // LDS float atomic (ds_add_f32)
    }
    // serial top-8 over 64
    float tp[TOPK_K];
    int   ti[TOPK_K];
    float tsum = 0.0f;
#pragma unroll
    for (int k = 0; k < TOPK_K; ++k) {
      float best = -1.0f;
      int   bi   = 0;
      for (int e = 0; e < E_SZ; ++e) {
        float p = lg[r][e];
        if (p > best) { best = p; bi = e; }
      }
      lg[r][bi] = -1.0f;
      tp[k] = best;
      ti[k] = bi;
      tsum += best;
    }
    const float rinv = 1.0f / (tsum + 1e-8f);
#pragma unroll
    for (int k = 0; k < TOPK_K; ++k) {
      out_topkp[m * TOPK_K + k] = tp[k] * rinv;
      out_topki[m * TOPK_K + k] = ti[k];
    }
  }
  __syncthreads();
  if (tid < E_SZ) atomicAdd(&usage_acc[tid], us[tid]);
}

// ---------------------------------------------------------------------------
__global__ void finalize_kernel(const float* __restrict__ usage_acc,
                                float* __restrict__ out_usage,
                                float* __restrict__ out_loss) {
  __shared__ float red[E_SZ];
  const int tid = threadIdx.x;
  float u = usage_acc[tid] * (1.0f / (float)B_SZ);
  out_usage[tid] = u;
  const float uni = 1.0f / (float)E_SZ;
  red[tid] = uni * (__logf(uni) - __logf(u));
  __syncthreads();
  if (tid == 0) {
    float s = 0.0f;
    for (int e = 0; e < E_SZ; ++e) s += red[e];
    out_loss[0] = s / (float)E_SZ;
  }
}

// ---------------------------------------------------------------------------
extern "C" void kernel_launch(void* const* d_in, const int* in_sizes, int n_in,
                              void* d_out, int out_size, void* d_ws,
                              size_t ws_size, hipStream_t stream) {
  (void)in_sizes; (void)n_in; (void)out_size; (void)ws_size;

  const float* x    = (const float*)d_in[0];
  const float* ctx  = (const float*)d_in[1];
  const float* W_ih = (const float*)d_in[2];
  const float* W_hh = (const float*)d_in[3];
  const float* b_ih = (const float*)d_in[4];
  const float* b_hh = (const float*)d_in[5];
  const float* fw1  = (const float*)d_in[6];
  const float* fb1  = (const float*)d_in[7];
  const float* fw2  = (const float*)d_in[8];
  const float* fb2  = (const float*)d_in[9];
  const float* gw1  = (const float*)d_in[10];
  const float* gb1  = (const float*)d_in[11];
  const float* gw2  = (const float*)d_in[12];
  const float* gb2  = (const float*)d_in[13];
  const float* spec = (const float*)d_in[14];

  // workspace layout:
  //   h16   [B*H]  f16     (zero-init)
  //   c_buf [B*H]  f32     (zero-init)
  //   usage [64]   f32     (zero-init)
  //   flow  [B*4]  f32
  //   wih16 [2048*1024] f16, whh16 [2048*512] f16, fw116 [512*1024] f16,
  //   gw116 [1024*1536] f16, gw216 [64*1024] f16
  _Float16* h16   = (_Float16*)d_ws;
  float*    c_buf = (float*)(h16 + (size_t)B_SZ * H_SZ);
  float*    usage = c_buf + (size_t)B_SZ * H_SZ;
  float*    flow  = usage + E_SZ;
  _Float16* wih16 = (_Float16*)(flow + (size_t)B_SZ * NFLOW);
  _Float16* whh16 = wih16 + (size_t)G4H * D_SZ;
  _Float16* fw116 = whh16 + (size_t)G4H * H_SZ;
  _Float16* gw116 = fw116 + (size_t)H_SZ * D_SZ;
  _Float16* gw216 = gw116 + (size_t)D_SZ * KW_G16;

  // output layout: topk_probs[B*8], topk_idx[B*8](int32 bits), loss[1],
  //                flow_state[B*4], expert_usage[64], adjusted_logits[B*64]
  float* out      = (float*)d_out;
  float* o_topkp  = out;
  int*   o_topki  = (int*)(out + (size_t)B_SZ * TOPK_K);
  float* o_loss   = out + 2ull * B_SZ * TOPK_K;
  float* o_flow   = o_loss + 1;
  float* o_usage  = o_flow + (size_t)B_SZ * NFLOW;
  float* o_adj    = o_usage + E_SZ;

  // zero h16 (as half of its float-equivalent span) + c + usage (contiguous)
  const unsigned long nz =
      (unsigned long)B_SZ * H_SZ / 2 + (unsigned long)B_SZ * H_SZ + E_SZ;
  zero_kernel<<<(unsigned)((nz + 255) / 256), 256, 0, stream>>>((float*)d_ws, nz);

  // weight conversions fp32 -> f16
  {
    long t;
    t = (long)G4H * D_SZ;
    cvt_kernel<<<(unsigned)((t + 255) / 256), 256, 0, stream>>>(W_ih, wih16, D_SZ, D_SZ, D_SZ, t);
    t = (long)G4H * H_SZ;
    cvt_kernel<<<(unsigned)((t + 255) / 256), 256, 0, stream>>>(W_hh, whh16, H_SZ, H_SZ, H_SZ, t);
    t = (long)H_SZ * D_SZ;
    cvt_kernel<<<(unsigned)((t + 255) / 256), 256, 0, stream>>>(fw1, fw116, D_SZ, D_SZ, D_SZ, t);
    t = (long)D_SZ * KW_G16;
    cvt_kernel<<<(unsigned)((t + 255) / 256), 256, 0, stream>>>(gw1, gw116, KW_GW1, KW_G16, KW_G16, t);
    t = (long)E_SZ * D_SZ;
    cvt_kernel<<<(unsigned)((t + 255) / 256), 256, 0, stream>>>(gw2, gw216, D_SZ, D_SZ, D_SZ, t);
  }

  for (int t = 0; t < T_SZ; ++t) {
    lstm_step_kernel<<<B_SZ / 32, 256, 0, stream>>>(ctx, wih16, whh16, b_ih,
                                                    b_hh, h16, c_buf, t);
  }
  flow_kernel<<<B_SZ / 16, 256, 0, stream>>>(x, fw116, fb1, fw2, fb2, flow,
                                             o_flow);
  gate_kernel<<<B_SZ / 16, 256, 0, stream>>>(x, h16, flow, gw116, gw1, gb1,
                                             gw216, gb2, spec, usage, o_topkp,
                                             o_topki, o_adj);
  finalize_kernel<<<1, E_SZ, 0, stream>>>(usage, o_usage, o_loss);
}
